// StandardAttention_35622458753867
// MI455X (gfx1250) — matrix operand order repaired and checked by reading, not compiled, not run on hardware
//
#include <hip/hip_runtime.h>
#include <stdint.h>

typedef __attribute__((ext_vector_type(16))) __bf16 v16bf;
typedef __attribute__((ext_vector_type(8)))  __bf16 v8bf;
typedef __attribute__((ext_vector_type(8)))  float  v8f;

#define CAT16(lo, hi) __builtin_shufflevector((lo), (hi), 0,1,2,3,4,5,6,7,8,9,10,11,12,13,14,15)

static __device__ __forceinline__ int iclampi(int v, int lo, int hi) {
    return v < lo ? lo : (v > hi ? hi : v);
}

// ---------------------------------------------------------------------------
// Fold conv bias + BatchNorm (eval) into per-channel scale/shift:
//   y = conv*S + T,  S = bw/sqrt(bv+eps),  T = bias*S + bb - bm*S
// ---------------------------------------------------------------------------
__global__ void k_fuse(const float* __restrict__ bias,
                       const float* __restrict__ bw, const float* __restrict__ bb,
                       const float* __restrict__ bm, const float* __restrict__ bv,
                       float* __restrict__ S, float* __restrict__ T, int C, int hasBN)
{
    for (int c = threadIdx.x; c < C; c += blockDim.x) {
        float s = 1.f;
        float t = bias ? bias[c] : 0.f;
        if (hasBN) {
            s = bw[c] * rsqrtf(bv[c] + 1e-5f);
            t = t * s + bb[c] - bm[c] * s;
        }
        S[c] = s; T[c] = t;
    }
}

// ---------------------------------------------------------------------------
// Repack OIHW fp32 weights -> bf16 GEMM A matrix [Cout][Ktot], k=(ky*3+kx)*Cin+ci
// ---------------------------------------------------------------------------
__global__ void k_repack(const float* __restrict__ w, __bf16* __restrict__ o,
                         int Cin, int Cout, int Ktot)
{
    int idx = blockIdx.x * blockDim.x + threadIdx.x;
    if (idx >= Cout * Ktot) return;
    int co = idx / Ktot, k = idx % Ktot;
    int t = k / Cin, ci = k % Cin;
    int ky = t / 3, kx = t % 3;
    o[idx] = (__bf16)w[((co * Cin + ci) * 3 + ky) * 3 + kx];
}

// ---------------------------------------------------------------------------
// combined = concat([x, resize(fused, 2x)], ch)  NCHW fp32 -> NHWC bf16 (128 ch)
// ---------------------------------------------------------------------------
__global__ void k_concat(const float* __restrict__ x, const float* __restrict__ fu,
                         __bf16* __restrict__ out)
{
    int idx = blockIdx.x * blockDim.x + threadIdx.x;   // 16384 px * 128 ch
    if (idx >= 16384 * 128) return;
    int c = idx & 127, p = idx >> 7;
    int b = p >> 12, rem = p & 4095, y = rem >> 6, xx = rem & 63;
    float val;
    if (c < 64) {
        val = x[(((size_t)(b * 64 + c)) << 12) + (y << 6) + xx];
    } else {
        int cf = c - 64;                    // bilinear 32->64, half-pixel centers
        float sy = (y + 0.5f) * 0.5f - 0.5f;
        float sx = (xx + 0.5f) * 0.5f - 0.5f;
        int y0 = (int)floorf(sy), x0 = (int)floorf(sx);
        float fy = sy - y0, fx = sx - x0;
        int y0c = iclampi(y0, 0, 31), y1c = iclampi(y0 + 1, 0, 31);
        int x0c = iclampi(x0, 0, 31), x1c = iclampi(x0 + 1, 0, 31);
        size_t base = ((size_t)(b * 64 + cf)) << 10;
        float g00 = fu[base + (y0c << 5) + x0c];
        float g01 = fu[base + (y0c << 5) + x1c];
        float g10 = fu[base + (y1c << 5) + x0c];
        float g11 = fu[base + (y1c << 5) + x1c];
        val = (g00 * (1.f - fx) + g01 * fx) * (1.f - fy) +
              (g10 * (1.f - fx) + g11 * fx) * fy;
    }
    out[(size_t)idx] = (__bf16)val;
}

// ---------------------------------------------------------------------------
// 3x3 conv, pad 1, implicit GEMM with v_wmma_f32_16x16x32_bf16
// in: NHWC bf16.  wA: [Cout][Ktot] bf16.  out: NHWC bf16 (+S/T/act epilogue)
// block = 256 thr (8 waves), tile = 32 Cout x 64 px, wave tile = 16x16.
// K-chunks of 32 never cross a 3x3 tap (32 | Cin), so the tap is chunk-uniform
// and B staging is one predicated global_load_b128 per thread.
// Fragments are contiguous 16B runs in LDS -> ds_load_b128 pairs.
// ---------------------------------------------------------------------------
__global__ __launch_bounds__(256) void k_conv3(
    const __bf16* __restrict__ in, const __bf16* __restrict__ wA,
    const float* __restrict__ S, const float* __restrict__ T,
    const float* __restrict__ alpha, int act,
    __bf16* __restrict__ out,
    int H, int W, int lw, int lhw, int lc, int Cin, int Cout, int Ktot)
{
    __shared__ __align__(16) __bf16 At[32][40];    // [cout][k]  (pitch 40 = 80B)
    __shared__ __align__(16) __bf16 BtT[64][40];   // [px][k]    (transposed!)
    const int tid  = threadIdx.x;
    const int lane = tid & 31, wv = tid >> 5;
    const int pxbase = blockIdx.x << 6;
    const int cobase = blockIdx.y << 5;
    const int mtile  = wv >> 2;        // 0..1 (Cout sub-tile)
    const int ntile  = wv & 3;         // 0..3 (pixel sub-tile)

    // B-staging coords (one b128 of 8 channels per thread)
    const int s_px = tid >> 2;                 // 0..63
    const int s_cg = (tid & 3) << 3;           // 0,8,16,24
    const int pg   = pxbase + s_px;
    const int bN   = pg >> lhw;
    const int prem = pg & ((1 << lhw) - 1);
    const int py   = prem >> lw, pxx = prem & (W - 1);
    // A-staging coords
    const int a_r  = tid >> 2;
    const int a_cg = (tid & 3) << 3;

    v8f acc = {};
    const int nchunks = Ktot >> 5;
    for (int kc = 0; kc < nchunks; ++kc) {
        const int k0 = kc << 5;
        const int t9 = k0 >> lc;               // tap, uniform over chunk
        const int cibase = k0 & (Cin - 1);
        const int dy = t9 / 3 - 1, dx = t9 % 3 - 1;
        if (a_r < 32) {
            *(v8bf*)&At[a_r][a_cg] =
                *(const v8bf*)(wA + (size_t)(cobase + a_r) * Ktot + k0 + a_cg);
        }
        {
            const int yy = py + dy, xx = pxx + dx;
            v8bf bv = {};
            if ((unsigned)yy < (unsigned)H && (unsigned)xx < (unsigned)W) {
                const int pin = (bN << lhw) + (yy << lw) + xx;
                bv = *(const v8bf*)(in + (size_t)pin * Cin + cibase + s_cg);
            }
            *(v8bf*)&BtT[s_px][s_cg] = bv;
        }
        __syncthreads();
        // A fragment: lane L: M = L&15, K pairs {khA..khA+7} and {16+khA..}
        const int m   = (mtile << 4) + (lane & 15);
        const int khA = (lane >> 4) << 3;      // 0 or 8
        v8bf alo = *(const v8bf*)&At[m][khA];
        v8bf ahi = *(const v8bf*)&At[m][16 + khA];
        v16bf afrag = CAT16(alo, ahi);
        // B fragment: lane L: N = L&15, K pairs {khB..khB+7} and {16+khB..}
        const int n   = (ntile << 4) + (lane & 15);
        const int khB = (lane >> 4) << 3;      // 0 or 8
        v8bf blo = *(const v8bf*)&BtT[n][khB];
        v8bf bhi = *(const v8bf*)&BtT[n][16 + khB];
        v16bf bfrag = CAT16(blo, bhi);
        acc = __builtin_amdgcn_wmma_f32_16x16x32_bf16(
            false, afrag, false, bfrag, (short)0, acc, false, false);
        __syncthreads();
    }
    // epilogue: lane L: pixel = N = L&15; channels co0..co0+7 contiguous
    const float al = (act == 2) ? alpha[0] : 0.f;
    const int pgo = pxbase + (ntile << 4) + (lane & 15);
    const int co0 = cobase + (mtile << 4) + ((lane >> 4) << 3);
    v8bf ov;
#pragma unroll
    for (int r = 0; r < 8; ++r) {
        float v = acc[r] * S[co0 + r] + T[co0 + r];
        if (act == 1)      v = fmaxf(v, 0.f);
        else if (act == 2) v = v >= 0.f ? v : al * v;
        ov[r] = (__bf16)v;
    }
    *(v8bf*)(out + (size_t)pgo * Cout + co0) = ov;
}

// ---------------------------------------------------------------------------
// 2x2 max pool, NHWC bf16
// ---------------------------------------------------------------------------
__global__ void k_pool(const __bf16* __restrict__ in, __bf16* __restrict__ out,
                       int lc, int lw_o, int lhw_o, int lw_i, int lhw_i, int total)
{
    int idx = blockIdx.x * blockDim.x + threadIdx.x;
    if (idx >= total) return;
    int C = 1 << lc;
    int c = idx & (C - 1), p = idx >> lc;
    int b = p >> lhw_o, rem = p & ((1 << lhw_o) - 1);
    int y = rem >> lw_o, x = rem & ((1 << lw_o) - 1);
    int Wi = 1 << lw_i;
    int pin = (b << lhw_i) + ((2 * y) << lw_i) + 2 * x;
    float v0 = (float)in[(size_t)pin * C + c];
    float v1 = (float)in[(size_t)(pin + 1) * C + c];
    float v2 = (float)in[(size_t)(pin + Wi) * C + c];
    float v3 = (float)in[(size_t)(pin + Wi + 1) * C + c];
    out[(size_t)p * C + c] = (__bf16)fmaxf(fmaxf(v0, v1), fmaxf(v2, v3));
}

// ---------------------------------------------------------------------------
// bilinear resize (half-pixel centers), NHWC bf16
// ---------------------------------------------------------------------------
__global__ void k_resize(const __bf16* __restrict__ in, __bf16* __restrict__ out,
                         int lc, int Hin, int Win, int lw_i, int lhw_i,
                         int Hout, int Wout, int lw_o, int lhw_o, int total)
{
    int idx = blockIdx.x * blockDim.x + threadIdx.x;
    if (idx >= total) return;
    int C = 1 << lc;
    int c = idx & (C - 1), p = idx >> lc;
    int b = p >> lhw_o, rem = p & ((1 << lhw_o) - 1);
    int oy = rem >> lw_o, ox = rem & ((1 << lw_o) - 1);
    float sh = (float)Hin / (float)Hout, sw = (float)Win / (float)Wout;
    float sy = (oy + 0.5f) * sh - 0.5f;
    float sx = (ox + 0.5f) * sw - 0.5f;
    int y0 = (int)floorf(sy), x0 = (int)floorf(sx);
    float fy = sy - y0, fx = sx - x0;
    int y0c = iclampi(y0, 0, Hin - 1), y1c = iclampi(y0 + 1, 0, Hin - 1);
    int x0c = iclampi(x0, 0, Win - 1), x1c = iclampi(x0 + 1, 0, Win - 1);
    size_t base = (size_t)b << lhw_i;
    float g00 = (float)in[(base + ((size_t)y0c << lw_i) + x0c) * C + c];
    float g01 = (float)in[(base + ((size_t)y0c << lw_i) + x1c) * C + c];
    float g10 = (float)in[(base + ((size_t)y1c << lw_i) + x0c) * C + c];
    float g11 = (float)in[(base + ((size_t)y1c << lw_i) + x1c) * C + c];
    float v = (g00 * (1.f - fx) + g01 * fx) * (1.f - fy) +
              (g10 * (1.f - fx) + g11 * fx) * fy;
    out[(size_t)p * C + c] = (__bf16)v;
}

// ---------------------------------------------------------------------------
// 1x1 conv. in: NHWC bf16. out: channel-major [b][co][n] fp32 and/or bf16.
// ---------------------------------------------------------------------------
__global__ void k_conv1(const __bf16* __restrict__ in, const float* __restrict__ w,
                        const float* __restrict__ bias,
                        float* __restrict__ outF, __bf16* __restrict__ outB,
                        int Cin, int Cout, int Npb, int total)
{
    int idx = blockIdx.x * blockDim.x + threadIdx.x;
    if (idx >= total) return;
    int co = idx % Cout, p = idx / Cout;
    float acc = bias[co];
    const __bf16* ip = in + (size_t)p * Cin;
    const float*  wp = w + (size_t)co * Cin;
    for (int ci = 0; ci < Cin; ci += 8) {
        v8bf xv = *(const v8bf*)(ip + ci);
#pragma unroll
        for (int u = 0; u < 8; ++u) acc += (float)xv[u] * wp[ci + u];
    }
    int b = p / Npb, n = p % Npb;
    size_t o = ((size_t)(b * Cout + co)) * Npb + n;
    if (outF) outF[o] = acc;
    if (outB) outB[o] = (__bf16)acc;
}

// ---------------------------------------------------------------------------
// Position attention. One block = (batch b, 16 query rows i).
// q,k fp32 [b][8][4096]; v bf16 [b][64][4096]; y1 residual NHWC bf16.
// Energy stored bf16 in 128KB dynamic LDS (padded pitch for conflict-free
// b128 row reads). Row max tracked in registers during the energy pass;
// unnormalized exp(e-max) written in place (threads own disjoint elements);
// 1/sum folded into the WMMA epilogue (softmax norm commutes with V@P^T).
// ---------------------------------------------------------------------------
#define PAM_PITCH 4104   // 4096 + 8 bf16 -> 8208B rows, 16B aligned, bank-spread
__global__ __launch_bounds__(256) void k_pam(
    const float* __restrict__ q, const float* __restrict__ kk,
    const __bf16* __restrict__ v, const __bf16* __restrict__ y1,
    const float* __restrict__ gammap, __bf16* __restrict__ out)
{
    extern __shared__ char pam_lds[];
    __bf16* Eb = (__bf16*)pam_lds;          // 16 * PAM_PITCH bf16 (~128 KB)
    __shared__ float qs[16][8];
    __shared__ float red[256];
    __shared__ float rmax[16], rsinv[16];
    const int tid = threadIdx.x;
    const int b = blockIdx.y;
    const int ibase = blockIdx.x << 4;

    if (tid < 128) {
        int i = tid >> 3, c = tid & 7;
        qs[i][c] = q[(((size_t)(b * 8 + c)) << 12) + ibase + i];
    }
    __syncthreads();
    // energy pass: thread (i = tid>>4) computes 32 chunks of 8 j's, tracks max
    const int i = tid >> 4, l16 = tid & 15;
    const float* kbase = kk + (((size_t)(b * 8)) << 12);
    float pmax = -3.4e38f;
    for (int s = 0; s < 32; ++s) {
        const int jb = (l16 << 3) + (s << 7);
        float vals[8];
#pragma unroll
        for (int u = 0; u < 8; ++u) vals[u] = 0.f;
#pragma unroll
        for (int c = 0; c < 8; ++c) {
            const float qv = qs[i][c];
            const float* kp = kbase + ((size_t)c << 12) + jb;
#pragma unroll
            for (int u = 0; u < 8; ++u) vals[u] += qv * kp[u];
        }
        v8bf ev;
#pragma unroll
        for (int u = 0; u < 8; ++u) { pmax = fmaxf(pmax, vals[u]); ev[u] = (__bf16)vals[u]; }
        *(v8bf*)&Eb[i * PAM_PITCH + jb] = ev;
    }
    red[tid] = pmax;
    __syncthreads();
    if (tid < 16) {
        float mm = -3.4e38f;
        for (int s = 0; s < 16; ++s) mm = fmaxf(mm, red[tid * 16 + s]);
        rmax[tid] = mm;
    }
    __syncthreads();
    // unnormalized probs in place + row sum
    const float rm = rmax[i];
    float ssum = 0.f;
    for (int s = 0; s < 32; ++s) {
        const int jb = (l16 << 3) + (s << 7);
        v8bf ev = *(const v8bf*)&Eb[i * PAM_PITCH + jb];
        v8bf pv;
#pragma unroll
        for (int u = 0; u < 8; ++u) {
            float p = __expf((float)ev[u] - rm);
            ssum += p;
            pv[u] = (__bf16)p;
        }
        *(v8bf*)&Eb[i * PAM_PITCH + jb] = pv;
    }
    red[tid] = ssum;
    __syncthreads();
    if (tid < 16) {
        float s = 0.f;
        for (int t = 0; t < 16; ++t) s += red[tid * 16 + t];
        rsinv[tid] = 1.f / s;
    }
    __syncthreads();
    // out[c,i] = rsinv[i] * sum_j v[c,j] * P[i,j]  -- WMMA, waves split K range
    const int lane = tid & 31, wv = tid >> 5;
    const int ctile = wv & 3, jhalf = wv >> 2;
    v8f acc = {};
    const int cch = (ctile << 4) + (lane & 15);
    const __bf16* vrow = v + (((size_t)(b * 64 + cch)) << 12);
    const int khA = (lane >> 4) << 3;
    const int irow = lane & 15;
    const int khB = (lane >> 4) << 3;
    for (int jc = 0; jc < 64; ++jc) {
        const int jb = (jhalf << 11) + (jc << 5);
        v8bf alo = *(const v8bf*)(vrow + jb + khA);
        v8bf ahi = *(const v8bf*)(vrow + jb + 16 + khA);
        v16bf afrag = CAT16(alo, ahi);
        v8bf blo = *(const v8bf*)&Eb[irow * PAM_PITCH + jb + khB];
        v8bf bhi = *(const v8bf*)&Eb[irow * PAM_PITCH + jb + khB + 16];
        v16bf bfrag = CAT16(blo, bhi);
        acc = __builtin_amdgcn_wmma_f32_16x16x32_bf16(
            false, afrag, false, bfrag, (short)0, acc, false, false);
    }
    __syncthreads();
    float* redt = (float*)pam_lds;           // probs dead; reuse for reduction
    if (jhalf == 1) {
#pragma unroll
        for (int r = 0; r < 8; ++r) redt[ctile * 256 + r * 32 + lane] = acc[r];
    }
    __syncthreads();
    if (jhalf == 0) {
        const float gamma = gammap[0];
        const int iloc = lane & 15;
        const int n = ibase + iloc;
        const size_t p = (((size_t)b) << 12) + n;
        const int co0 = (ctile << 4) + ((lane >> 4) << 3);
        const float ns = rsinv[iloc];
        v8bf res = *(const v8bf*)(y1 + p * 64 + co0);
        v8bf ov;
#pragma unroll
        for (int r = 0; r < 8; ++r) {
            float s = (acc[r] + redt[ctile * 256 + r * 32 + lane]) * ns;
            ov[r] = (__bf16)(gamma * s + (float)res[r]);
        }
        *(v8bf*)(out + p * 64 + co0) = ov;
    }
}

// ---------------------------------------------------------------------------
// Channel attention (one block per batch). xf NHWC bf16 (4096 px, 64 ch).
// energy[c,d] = sum_n xf[c,n]xf[d,n]; e' = rowmax - e; softmax;
// out = gamma * attn@xf + xf.
// ---------------------------------------------------------------------------
__global__ __launch_bounds__(256) void k_cam(
    const __bf16* __restrict__ xf, const float* __restrict__ gammap,
    __bf16* __restrict__ out)
{
    __shared__ __bf16 xt[64][130];
    __shared__ float attn[64][65];
    const int tid = threadIdx.x;
    const int b = blockIdx.x;
    float acc[16];
#pragma unroll
    for (int qd = 0; qd < 16; ++qd) acc[qd] = 0.f;

    for (int nc = 0; nc < 32; ++nc) {
        __syncthreads();
        for (int it = 0; it < 32; ++it) {
            int e = tid + (it << 8);
            int c = e & 63, nl = e >> 6;
            xt[c][nl] = xf[((((size_t)b) << 12) + (nc << 7) + nl) * 64 + c];
        }
        __syncthreads();
#pragma unroll
        for (int qd = 0; qd < 16; ++qd) {
            int pid = tid * 16 + qd;
            int c = pid >> 6, d = pid & 63;
            float s = 0.f;
            for (int nl = 0; nl < 128; ++nl)
                s += (float)xt[c][nl] * (float)xt[d][nl];
            acc[qd] += s;
        }
    }
    __syncthreads();
#pragma unroll
    for (int qd = 0; qd < 16; ++qd) {
        int pid = tid * 16 + qd;
        attn[pid >> 6][pid & 63] = acc[qd];
    }
    __syncthreads();
    if (tid < 64) {
        float mx = -3.4e38f, mn = 3.4e38f;
        for (int d = 0; d < 64; ++d) {
            float e = attn[tid][d];
            mx = fmaxf(mx, e); mn = fminf(mn, e);
        }
        float vm = mx - mn;                    // max of (mx - e)
        float denom = 0.f;
        for (int d = 0; d < 64; ++d) denom += __expf((mx - attn[tid][d]) - vm);
        float inv = 1.f / denom;
        for (int d = 0; d < 64; ++d)
            attn[tid][d] = __expf((mx - attn[tid][d]) - vm) * inv;
    }
    __syncthreads();
    const float gamma = gammap[0];
    for (int nc = 0; nc < 32; ++nc) {
        __syncthreads();
        for (int it = 0; it < 32; ++it) {
            int e = tid + (it << 8);
            int c = e & 63, nl = e >> 6;
            xt[c][nl] = xf[((((size_t)b) << 12) + (nc << 7) + nl) * 64 + c];
        }
        __syncthreads();
        for (int qd = 0; qd < 32; ++qd) {
            int pid = tid + (qd << 8);
            int c = pid & 63, nl = pid >> 6;
            float s = 0.f;
            for (int d = 0; d < 64; ++d) s += attn[c][d] * (float)xt[d][nl];
            size_t p = (((size_t)b) << 12) + (nc << 7) + nl;
            float val = gamma * s + (float)xt[c][nl];
            out[p * 64 + c] = (__bf16)val;
        }
    }
}

// ---------------------------------------------------------------------------
// t = (a + b) * s  (bf16 NHWC elementwise)
// ---------------------------------------------------------------------------
__global__ void k_comb(const __bf16* __restrict__ a, const __bf16* __restrict__ b,
                       const __bf16* __restrict__ s, __bf16* __restrict__ o, int total)
{
    int idx = blockIdx.x * blockDim.x + threadIdx.x;
    if (idx >= total) return;
    o[idx] = (__bf16)(((float)a[idx] + (float)b[idx]) * (float)s[idx]);
}

// ===========================================================================
extern "C" void kernel_launch(void* const* d_in, const int* in_sizes, int n_in,
                              void* d_out, int out_size, void* d_ws, size_t ws_size,
                              hipStream_t stream)
{
    (void)in_sizes; (void)n_in; (void)out_size; (void)ws_size;
    auto f = [&](int i) { return (const float*)d_in[i]; };

    char* ws = (char*)d_ws;
    size_t off = 0;
    auto alloc = [&](size_t bytes) -> void* {
        void* p = ws + off;
        off = (off + bytes + 255) & ~(size_t)255;
        return p;
    };

    // ---- activation buffers (NHWC bf16 unless noted) ----
    __bf16* combined = (__bf16*)alloc((size_t)16384 * 128 * 2);
    __bf16* e1a = (__bf16*)alloc((size_t)16384 * 256 * 2);
    __bf16* e1b = (__bf16*)alloc((size_t)16384 * 256 * 2);
    __bf16* e1p = (__bf16*)alloc((size_t)4096 * 256 * 2);
    __bf16* e2a = (__bf16*)alloc((size_t)4096 * 512 * 2);
    __bf16* e2b = (__bf16*)alloc((size_t)4096 * 512 * 2);
    __bf16* e2p = (__bf16*)alloc((size_t)1024 * 512 * 2);
    __bf16* d2a = (__bf16*)alloc((size_t)4096 * 256 * 2);
    __bf16* d2b = (__bf16*)alloc((size_t)4096 * 256 * 2);
    __bf16* y1p = (__bf16*)alloc((size_t)16384 * 64 * 2);
    __bf16* y1c = (__bf16*)alloc((size_t)16384 * 64 * 2);
    float*  qb  = (float*)alloc((size_t)4 * 8 * 4096 * 4);
    float*  kb  = (float*)alloc((size_t)4 * 8 * 4096 * 4);
    __bf16* vb  = (__bf16*)alloc((size_t)4 * 64 * 4096 * 2);
    __bf16* pam_o = (__bf16*)alloc((size_t)16384 * 64 * 2);
    __bf16* pam2  = (__bf16*)alloc((size_t)16384 * 64 * 2);
    __bf16* cam_o = (__bf16*)alloc((size_t)16384 * 64 * 2);
    __bf16* cam2  = (__bf16*)alloc((size_t)16384 * 64 * 2);
    __bf16* sem   = (__bf16*)alloc((size_t)16384 * 64 * 2);
    __bf16* comb2 = (__bf16*)alloc((size_t)16384 * 64 * 2);
    // aliases for dead buffers (decoder)
    __bf16* d2up = e2a;                 // 4096*512, e2a dead after e2c2
    __bf16* d1up = e1a;                 // 16384*256, e1a dead after e1c2
    __bf16* d1a  = e1b;                 // 16384*128 (fits in dead e1b)
    __bf16* d1b  = e1b + (size_t)16384 * 128;

    // ---- conv3x3 launcher: fuse BN, repack weights, run WMMA conv ----
    auto conv3 = [&](const __bf16* in, int H, int W, int Cin, int Cout,
                     const float* w, const float* bias,
                     const float* bw, const float* bb,
                     const float* bm, const float* bv,
                     int act, const float* alpha, __bf16* out) {
        int Ktot = 9 * Cin;
        float* S = (float*)alloc((size_t)Cout * 4);
        float* T = (float*)alloc((size_t)Cout * 4);
        __bf16* wA = (__bf16*)alloc((size_t)Cout * Ktot * 2);
        k_fuse<<<1, 256, 0, stream>>>(bias, bw, bb, bm, bv, S, T, Cout, bw != nullptr);
        int tot = Cout * Ktot;
        k_repack<<<(tot + 255) / 256, 256, 0, stream>>>(w, wA, Cin, Cout, Ktot);
        int P = 4 * H * W;
        dim3 g(P >> 6, Cout >> 5);
        k_conv3<<<g, 256, 0, stream>>>(in, wA, S, T, alpha, act, out, H, W,
                                       __builtin_ctz(W), __builtin_ctz(H * W),
                                       __builtin_ctz(Cin), Cin, Cout, Ktot);
    };

    const float* X  = f(0);
    const float* FU = f(1);

    // combined = concat(x, resize2x(fused))
    k_concat<<<(16384 * 128 + 255) / 256, 256, 0, stream>>>(X, FU, combined);

    // ---- SemanticModule ----
    conv3(combined, 64, 64, 128, 256, f(2),  f(3),  f(4),  f(5),  f(6),  f(7),  1, nullptr, e1a);
    conv3(e1a,      64, 64, 256, 256, f(8),  f(9),  f(10), f(11), f(12), f(13), 1, nullptr, e1b);
    k_pool<<<(1048576 + 255) / 256, 256, 0, stream>>>(e1b, e1p, 8, 5, 10, 6, 12, 1048576);
    conv3(e1p,      32, 32, 256, 512, f(14), f(15), f(16), f(17), f(18), f(19), 1, nullptr, e2a);
    conv3(e2a,      32, 32, 512, 512, f(20), f(21), f(22), f(23), f(24), f(25), 1, nullptr, e2b);
    k_pool<<<(524288 + 255) / 256, 256, 0, stream>>>(e2b, e2p, 9, 4, 8, 5, 10, 524288);
    k_resize<<<(2097152 + 255) / 256, 256, 0, stream>>>(e2p, d2up, 9, 16, 16, 4, 8, 32, 32, 5, 10, 2097152);
    conv3(d2up,     32, 32, 512, 256, f(26), f(27), f(28), f(29), f(30), f(31), 1, nullptr, d2a);
    conv3(d2a,      32, 32, 256, 256, f(32), f(33), f(34), f(35), f(36), f(37), 1, nullptr, d2b);
    // d2 resize to e1 pooled size is identity (32x32 -> 32x32)
    k_resize<<<(4194304 + 255) / 256, 256, 0, stream>>>(d2b, d1up, 8, 32, 32, 5, 10, 64, 64, 6, 12, 4194304);
    conv3(d1up,     64, 64, 256, 128, f(38), f(39), f(40), f(41), f(42), f(43), 1, nullptr, d1a);
    conv3(d1a,      64, 64, 128, 128, f(44), f(45), f(46), f(47), f(48), f(49), 1, nullptr, d1b);

    // ---- PAM branch ----
    conv3(combined, 64, 64, 128, 64, f(50), f(51), f(52), f(53), f(54), f(55), 2, f(56), y1p);
    k_conv1<<<(131072 + 255) / 256, 256, 0, stream>>>(y1p, f(57), f(58), qb, nullptr, 64, 8, 4096, 131072);
    k_conv1<<<(131072 + 255) / 256, 256, 0, stream>>>(y1p, f(59), f(60), kb, nullptr, 64, 8, 4096, 131072);
    k_conv1<<<(1048576 + 255) / 256, 256, 0, stream>>>(y1p, f(61), f(62), nullptr, vb, 64, 64, 4096, 1048576);
    k_pam<<<dim3(256, 4), 256, 16 * PAM_PITCH * sizeof(__bf16), stream>>>(qb, kb, vb, y1p, f(63), pam_o);
    conv3(pam_o,    64, 64, 64, 64, f(64), f(65), f(66), f(67), f(68), f(69), 2, f(70), pam2);

    // ---- CAM branch ----
    conv3(combined, 64, 64, 128, 64, f(71), f(72), f(73), f(74), f(75), f(76), 2, f(77), y1c);
    k_cam<<<4, 256, 0, stream>>>(y1c, f(78), cam_o);
    conv3(cam_o,    64, 64, 64, 64, f(79), f(80), f(81), f(82), f(83), f(84), 2, f(85), cam2);

    // ---- head ----
    conv3(d1b,      64, 64, 128, 64, f(88), f(89), nullptr, nullptr, nullptr, nullptr, 0, nullptr, sem);
    k_comb<<<(1048576 + 255) / 256, 256, 0, stream>>>(pam2, cam2, sem, comb2, 1048576);
    k_conv1<<<(1048576 + 255) / 256, 256, 0, stream>>>(comb2, f(86), f(87), (float*)d_out, nullptr, 64, 64, 4096, 1048576);
}
